// dca_51376398795429
// MI455X (gfx1250) — matrix operand order repaired and checked by reading, not compiled, not run on hardware
//
// CDNA5 / gfx1250 DCA (ZINB autoencoder) forward pass.
// bf16 WMMA GEMMs with double-buffered LDS fed by CDNA5 async-to-LDS copies
// (global_load_async_to_lds_b128/b64 + s_wait_asynccnt), fused epilogues
// (bias+relu / sigmoid / exp / exp + s*M). BN + log1p handled by
// bandwidth-bound elementwise kernels; exact median via single-block rank
// selection. Everything launched on `stream`; scratch carved out of d_ws.

#include <hip/hip_runtime.h>
#include <hip/hip_bf16.h>

#define B_    2048
#define INDIM 20000
#define ENC_  512
#define BOT_  128
#define DEC_  512

typedef __attribute__((ext_vector_type(16))) __bf16 v16bf;
typedef __attribute__((ext_vector_type(8)))  __bf16 v8bf;
typedef __attribute__((ext_vector_type(8)))  float  v8f;

__device__ __forceinline__ __bf16 to_bf16(float f) {
  // round-to-nearest-even f32 -> bf16
  unsigned u = __float_as_uint(f);
  unsigned r = (u + 0x7FFFu + ((u >> 16) & 1u)) >> 16;
  unsigned short hs = (unsigned short)r;
  __bf16 b;
  __builtin_memcpy(&b, &hs, 2);
  return b;
}

// ---- CDNA5 async global->LDS copies (ASYNCcnt-tracked, no VGPR round trip)
__device__ __forceinline__ unsigned lds_off_of(const void* p) {
  // generic pointer to LDS: addr[31:0] is the LDS byte offset (ISA 10.2)
  return (unsigned)(size_t)p;
}
__device__ __forceinline__ void async_copy_b128(unsigned lds_byte_off, const void* gaddr) {
  asm volatile("global_load_async_to_lds_b128 %0, %1, off"
               :: "v"(lds_byte_off), "v"(gaddr)
               : "memory");
}
__device__ __forceinline__ void async_copy_b64(unsigned lds_byte_off, const void* gaddr) {
  asm volatile("global_load_async_to_lds_b64 %0, %1, off"
               :: "v"(lds_byte_off), "v"(gaddr)
               : "memory");
}
__device__ __forceinline__ void wait_async0() {
  asm volatile("s_wait_asynccnt 0x0" ::: "memory");
}

// ---------------------------------------------------------------------------
// 1) row sums of x: one block per row, 256-thread tree reduction
// ---------------------------------------------------------------------------
__global__ __launch_bounds__(256) void rowsum_kernel(const float* __restrict__ x,
                                                     float* __restrict__ row_sum,
                                                     int cols) {
  const int row = blockIdx.x;
  const float* xr = x + (size_t)row * cols;
  float acc = 0.0f;
  for (int c = threadIdx.x; c < cols; c += blockDim.x) acc += xr[c];
  __shared__ float sm[256];
  sm[threadIdx.x] = acc;
  __syncthreads();
  for (int off = 128; off > 0; off >>= 1) {
    if (threadIdx.x < off) sm[threadIdx.x] += sm[threadIdx.x + off];
    __syncthreads();
  }
  if (threadIdx.x == 0) row_sum[row] = sm[0];
}

// ---------------------------------------------------------------------------
// 2) exact median of 2048 row sums (avg of order stats 1023,1024) -> s, 1/s
//    Single block, stable-rank selection (2048^2 comparisons in LDS).
// ---------------------------------------------------------------------------
__global__ __launch_bounds__(1024) void median_s_kernel(const float* __restrict__ row_sum,
                                                        float* __restrict__ s,
                                                        float* __restrict__ inv_s) {
  __shared__ float v[B_];
  __shared__ float med[2];
  const int t = threadIdx.x;  // 1024 threads
  v[t] = row_sum[t];
  v[t + 1024] = row_sum[t + 1024];
  __syncthreads();
  for (int e = 0; e < 2; ++e) {
    const int i = t + e * 1024;
    const float xv = v[i];
    int rank = 0;
    for (int j = 0; j < B_; ++j) {
      const float y = v[j];
      rank += (y < xv) || (y == xv && j < i);
    }
    if (rank == 1023) med[0] = xv;
    if (rank == 1024) med[1] = xv;
  }
  __syncthreads();
  const float m = 0.5f * (med[0] + med[1]);
  for (int e = 0; e < 2; ++e) {
    const int i = t + e * 1024;
    const float sv = v[i] / m;
    s[i] = sv;
    inv_s[i] = 1.0f / sv;
  }
}

// ---------------------------------------------------------------------------
// 3) BN0 stats over columns of h0 = log1p(x / s): one thread per column
//    Produces fused affine: scale = g*rstd, shift = b - mean*scale
// ---------------------------------------------------------------------------
__global__ __launch_bounds__(256) void bn0_stats_kernel(const float* __restrict__ x,
                                                        const float* __restrict__ inv_s,
                                                        const float* __restrict__ g,
                                                        const float* __restrict__ b,
                                                        float* __restrict__ scale,
                                                        float* __restrict__ shift,
                                                        int rows, int cols) {
  const int c = blockIdx.x * blockDim.x + threadIdx.x;
  if (c >= cols) return;
  float sum = 0.0f, sq = 0.0f;
  for (int r = 0; r < rows; ++r) {
    const float v = log1pf(x[(size_t)r * cols + c] * inv_s[r]);
    sum += v;
    sq += v * v;
  }
  const float mean = sum / (float)rows;
  const float var = fmaxf(sq / (float)rows - mean * mean, 0.0f);
  const float sc = rsqrtf(var + 1e-5f) * g[c];
  scale[c] = sc;
  shift[c] = b[c] - mean * sc;
}

// 3b) apply BN0 and emit bf16 A-matrix for the encoder GEMM
__global__ __launch_bounds__(256) void bn0_apply_kernel(const float* __restrict__ x,
                                                        const float* __restrict__ inv_s,
                                                        const float* __restrict__ scale,
                                                        const float* __restrict__ shift,
                                                        __bf16* __restrict__ out,
                                                        int cols, size_t total) {
  const size_t i = (size_t)blockIdx.x * blockDim.x + threadIdx.x;
  if (i >= total) return;
  const int c = (int)(i % cols);
  const int r = (int)(i / cols);
  const float v = log1pf(x[i] * inv_s[r]);
  out[i] = to_bf16(v * scale[c] + shift[c]);
}

// ---------------------------------------------------------------------------
// 4) BN stats / apply for the f32 intermediate activations
// ---------------------------------------------------------------------------
__global__ __launch_bounds__(256) void bn_stats_kernel(const float* __restrict__ h,
                                                       const float* __restrict__ g,
                                                       const float* __restrict__ b,
                                                       float* __restrict__ scale,
                                                       float* __restrict__ shift,
                                                       int rows, int cols) {
  const int c = blockIdx.x * blockDim.x + threadIdx.x;
  if (c >= cols) return;
  float sum = 0.0f, sq = 0.0f;
  for (int r = 0; r < rows; ++r) {
    const float v = h[(size_t)r * cols + c];
    sum += v;
    sq += v * v;
  }
  const float mean = sum / (float)rows;
  const float var = fmaxf(sq / (float)rows - mean * mean, 0.0f);
  const float sc = rsqrtf(var + 1e-5f) * g[c];
  scale[c] = sc;
  shift[c] = b[c] - mean * sc;
}

__global__ __launch_bounds__(256) void bn_apply_bf16_kernel(const float* __restrict__ h,
                                                            const float* __restrict__ scale,
                                                            const float* __restrict__ shift,
                                                            __bf16* __restrict__ out,
                                                            int cols, size_t total) {
  const size_t i = (size_t)blockIdx.x * blockDim.x + threadIdx.x;
  if (i >= total) return;
  const int c = (int)(i % cols);
  out[i] = to_bf16(h[i] * scale[c] + shift[c]);
}

// one-time f32 -> bf16 weight conversion (weights are reused by 32 M-blocks,
// so pre-converting halves their HBM traffic in the GEMMs)
__global__ __launch_bounds__(256) void cvt_bf16_kernel(const float* __restrict__ in,
                                                       __bf16* __restrict__ out, size_t n) {
  const size_t i = (size_t)blockIdx.x * blockDim.x + threadIdx.x;
  if (i < n) out[i] = to_bf16(in[i]);
}

// ---------------------------------------------------------------------------
// 5) WMMA GEMM: C[M,N] = epi(A[M,K] @ W[N,K]^T + bias)
//    A, W bf16 row-major (contiguous K for both). Block tile 64x32x32,
//    256 threads = 8 wave32 waves, one v_wmma_f32_16x16x32_bf16 per wave per
//    K-step. Double-buffered LDS fed by global_load_async_to_lds_* with a
//    1-deep software pipeline; only sync is s_wait_asynccnt 0 + barrier.
//    No divergence anywhere (EXEC all-ones, WMMA requirement).
// ---------------------------------------------------------------------------
enum { EPI_RELU_F32 = 0, EPI_RELU_BF16 = 1, EPI_SIGMOID = 2, EPI_EXP = 3, EPI_EXP_MLOSS = 4 };

template <int EPI>
__global__ __launch_bounds__(256) void gemm_wmma_bf16(
    const __bf16* __restrict__ A,    // [M,K]
    const __bf16* __restrict__ W,    // [N,K]
    const float*  __restrict__ bias, // [N]
    float* __restrict__ outF, __bf16* __restrict__ outB,
    float* __restrict__ out2, const float* __restrict__ srow,
    int M, int N, int K) {
  constexpr int BM = 64, BN = 32, BK = 32, STR = 40;  // STR pads rows (80B) vs bank conflicts
  __shared__ __align__(16) __bf16 As[2][BM * STR];
  __shared__ __align__(16) __bf16 Ws[2][BN * STR];

  const int t = threadIdx.x;
  const int wave = t >> 5;
  const int lane = t & 31;  // wave32
  const int tileM = (wave & 3) * 16;
  const int tileN = (wave >> 2) * 16;
  const int bm = blockIdx.x * BM;
  const int bn = blockIdx.y * BN;

  // Uniform cooperative staging (all 256 threads, no divergence):
  //  A tile 64x32: 256 x 16B chunks (row t>>2, 8 elems at (t&3)*8)
  //  W tile 32x32: 256 x  8B chunks (row t>>3, 4 elems at (t&7)*4)
  const int arow = t >> 2, acol = (t & 3) * 8;
  const int wrow = t >> 3, wcol = (t & 7) * 4;
  const __bf16* pa = A + (size_t)(bm + arow) * K + acol;
  const __bf16* pw = W + (size_t)(bn + wrow) * K + wcol;

  const unsigned ldsA0 = lds_off_of(&As[0][arow * STR + acol]);
  const unsigned ldsA1 = lds_off_of(&As[1][arow * STR + acol]);
  const unsigned ldsW0 = lds_off_of(&Ws[0][wrow * STR + wcol]);
  const unsigned ldsW1 = lds_off_of(&Ws[1][wrow * STR + wcol]);

  v8f acc;
#pragma unroll
  for (int i = 0; i < 8; ++i) acc[i] = 0.0f;

  // fragment addressing per documented CDNA5 bf16 VGPR layouts:
  //  A (16x32, M on lanes): lanes<16 -> K {0..7,16..23}; lanes>=16 -> K {8..15,24..31}
  //  B (32x16, N on lanes): lanes<16 -> K {0..7,16..23}; lanes>=16 -> K {8..15,24..31}
  const int half = lane >> 4;
  const int aOff = (tileM + (lane & 15)) * STR + half * 8;
  const int bOff = (tileN + (lane & 15)) * STR + half * 8;

  // prologue: stage tile k0=0 into buffer 0
  async_copy_b128(ldsA0, pa);
  async_copy_b64(ldsW0, pw);

  int buf = 0;
  for (int k0 = 0; k0 < K; k0 += BK) {
    wait_async0();     // my async writes into buf are done
    __syncthreads();   // everyone's are done; prior reads of buf^1 retired

    if (k0 + BK < K) {  // stage next tile into the other buffer (overlaps WMMA)
      pa += BK;
      pw += BK;
      async_copy_b128(buf ? ldsA0 : ldsA1, pa);
      async_copy_b64(buf ? ldsW0 : ldsW1, pw);
    }

    const __bf16* aBase = (buf ? &As[1][0] : &As[0][0]) + aOff;
    const __bf16* bBase = (buf ? &Ws[1][0] : &Ws[0][0]) + bOff;
    const v8bf a_lo = *(const v8bf*)(aBase);
    const v8bf a_hi = *(const v8bf*)(aBase + 16);
    const v8bf b_lo = *(const v8bf*)(bBase);
    const v8bf b_hi = *(const v8bf*)(bBase + 16);
    v16bf a, b;
#pragma unroll
    for (int i = 0; i < 8; ++i) {
      a[i] = a_lo[i];
      a[8 + i] = a_hi[i];
      b[i] = b_lo[i];
      b[8 + i] = b_hi[i];
    }
    acc = __builtin_amdgcn_wmma_f32_16x16x32_bf16(false, a, false, b, (short)0, acc,
                                                  false, false);
    buf ^= 1;
  }

  // C/D layout: lanes 0-15 -> N=lane, M=j; lanes 16-31 -> N=lane-16, M=8+j
  const int col = bn + tileN + (lane & 15);
  const float bc = bias[col];
  const int r0 = bm + tileM + half * 8;
#pragma unroll
  for (int j = 0; j < 8; ++j) {
    const size_t idx = (size_t)(r0 + j) * N + col;
    const float v = acc[j] + bc;
    if (EPI == EPI_RELU_F32) {
      outF[idx] = fmaxf(v, 0.0f);
    } else if (EPI == EPI_RELU_BF16) {
      outB[idx] = to_bf16(fmaxf(v, 0.0f));
    } else if (EPI == EPI_SIGMOID) {
      outF[idx] = 1.0f / (1.0f + expf(-v));
    } else if (EPI == EPI_EXP) {
      outF[idx] = expf(v);
    } else {  // EPI_EXP_MLOSS: M and M_loss = s*M in one pass
      const float e = expf(v);
      outF[idx] = e;
      out2[idx] = srow[r0 + j] * e;
    }
  }
}

// ---------------------------------------------------------------------------
// host orchestration
// ---------------------------------------------------------------------------
static inline size_t cdiv(size_t a, size_t b) { return (a + b - 1) / b; }

extern "C" void kernel_launch(void* const* d_in, const int* in_sizes, int n_in,
                              void* d_out, int out_size, void* d_ws, size_t ws_size,
                              hipStream_t stream) {
  (void)in_sizes; (void)n_in; (void)out_size; (void)ws_size;

  const float* x     = (const float*)d_in[0];
  const float* bn0_g = (const float*)d_in[1];
  const float* bn0_b = (const float*)d_in[2];
  const float* enc_W = (const float*)d_in[3];
  const float* enc_b = (const float*)d_in[4];
  const float* bn1_g = (const float*)d_in[5];
  const float* bn1_b = (const float*)d_in[6];
  const float* bot_W = (const float*)d_in[7];
  const float* bot_b = (const float*)d_in[8];
  const float* bn2_g = (const float*)d_in[9];
  const float* bn2_b = (const float*)d_in[10];
  const float* dec_W = (const float*)d_in[11];
  const float* dec_b = (const float*)d_in[12];
  const float* o1_W  = (const float*)d_in[13];
  const float* o1_b  = (const float*)d_in[14];
  const float* o2_W  = (const float*)d_in[15];
  const float* o2_b  = (const float*)d_in[16];
  const float* o3_W  = (const float*)d_in[17];
  const float* o3_b  = (const float*)d_in[18];

  float* out = (float*)d_out;  // pai | M | theta | M_loss, each [B, INDIM] f32
  float* pai   = out;
  float* Mout  = out + (size_t)B_ * INDIM;
  float* theta = out + 2 * (size_t)B_ * INDIM;
  float* Mloss = out + 3 * (size_t)B_ * INDIM;

  // carve scratch (~175 MB total)
  char* w = (char*)d_ws;
  size_t off = 0;
  auto alloc = [&](size_t bytes) -> void* {
    void* p = w + off;
    off = (off + bytes + 255) & ~(size_t)255;
    return p;
  };
  float*  row_sum = (float*)alloc(B_ * 4);
  float*  s       = (float*)alloc(B_ * 4);
  float*  inv_s   = (float*)alloc(B_ * 4);
  float*  sc0     = (float*)alloc((size_t)INDIM * 4);
  float*  sh0     = (float*)alloc((size_t)INDIM * 4);
  __bf16* h0      = (__bf16*)alloc((size_t)B_ * INDIM * 2);
  __bf16* encWb   = (__bf16*)alloc((size_t)ENC_ * INDIM * 2);
  __bf16* botWb   = (__bf16*)alloc((size_t)BOT_ * ENC_ * 2);
  __bf16* decWb   = (__bf16*)alloc((size_t)DEC_ * BOT_ * 2);
  __bf16* o1Wb    = (__bf16*)alloc((size_t)INDIM * DEC_ * 2);
  __bf16* o2Wb    = (__bf16*)alloc((size_t)INDIM * DEC_ * 2);
  __bf16* o3Wb    = (__bf16*)alloc((size_t)INDIM * DEC_ * 2);
  float*  h1      = (float*)alloc((size_t)B_ * ENC_ * 4);
  __bf16* h1b     = (__bf16*)alloc((size_t)B_ * ENC_ * 2);
  float*  sc1     = (float*)alloc(ENC_ * 4);
  float*  sh1     = (float*)alloc(ENC_ * 4);
  float*  h2      = (float*)alloc((size_t)B_ * BOT_ * 4);
  __bf16* h2b     = (__bf16*)alloc((size_t)B_ * BOT_ * 2);
  float*  sc2     = (float*)alloc(BOT_ * 4);
  float*  sh2     = (float*)alloc(BOT_ * 4);
  __bf16* h3b     = (__bf16*)alloc((size_t)B_ * DEC_ * 2);

  // --- normalization pipeline ---
  rowsum_kernel<<<B_, 256, 0, stream>>>(x, row_sum, INDIM);
  median_s_kernel<<<1, 1024, 0, stream>>>(row_sum, s, inv_s);
  bn0_stats_kernel<<<(unsigned)cdiv(INDIM, 256), 256, 0, stream>>>(x, inv_s, bn0_g, bn0_b,
                                                                   sc0, sh0, B_, INDIM);
  {
    const size_t tot = (size_t)B_ * INDIM;
    bn0_apply_kernel<<<(unsigned)cdiv(tot, 256), 256, 0, stream>>>(x, inv_s, sc0, sh0, h0,
                                                                   INDIM, tot);
  }

  // --- weight f32 -> bf16 (one pass each) ---
  {
    const size_t nenc = (size_t)ENC_ * INDIM;
    cvt_bf16_kernel<<<(unsigned)cdiv(nenc, 256), 256, 0, stream>>>(enc_W, encWb, nenc);
    const size_t nbot = (size_t)BOT_ * ENC_;
    cvt_bf16_kernel<<<(unsigned)cdiv(nbot, 256), 256, 0, stream>>>(bot_W, botWb, nbot);
    const size_t ndec = (size_t)DEC_ * BOT_;
    cvt_bf16_kernel<<<(unsigned)cdiv(ndec, 256), 256, 0, stream>>>(dec_W, decWb, ndec);
    const size_t nhead = (size_t)INDIM * DEC_;
    cvt_bf16_kernel<<<(unsigned)cdiv(nhead, 256), 256, 0, stream>>>(o1_W, o1Wb, nhead);
    cvt_bf16_kernel<<<(unsigned)cdiv(nhead, 256), 256, 0, stream>>>(o2_W, o2Wb, nhead);
    cvt_bf16_kernel<<<(unsigned)cdiv(nhead, 256), 256, 0, stream>>>(o3_W, o3Wb, nhead);
  }

  // --- encoder: h1 = relu(h0 @ enc_W^T + enc_b)  [2048,20000]x[512,20000]^T ---
  gemm_wmma_bf16<EPI_RELU_F32><<<dim3(B_ / 64, ENC_ / 32), 256, 0, stream>>>(
      h0, encWb, enc_b, h1, nullptr, nullptr, nullptr, B_, ENC_, INDIM);
  bn_stats_kernel<<<(unsigned)cdiv(ENC_, 256), 256, 0, stream>>>(h1, bn1_g, bn1_b, sc1, sh1,
                                                                 B_, ENC_);
  bn_apply_bf16_kernel<<<(unsigned)cdiv((size_t)B_ * ENC_, 256), 256, 0, stream>>>(
      h1, sc1, sh1, h1b, ENC_, (size_t)B_ * ENC_);

  // --- bottleneck: h2 = relu(h1b @ bot_W^T + bot_b)  [2048,512]x[128,512]^T ---
  gemm_wmma_bf16<EPI_RELU_F32><<<dim3(B_ / 64, BOT_ / 32), 256, 0, stream>>>(
      h1b, botWb, bot_b, h2, nullptr, nullptr, nullptr, B_, BOT_, ENC_);
  bn_stats_kernel<<<(unsigned)cdiv(BOT_, 256), 256, 0, stream>>>(h2, bn2_g, bn2_b, sc2, sh2,
                                                                 B_, BOT_);
  bn_apply_bf16_kernel<<<(unsigned)cdiv((size_t)B_ * BOT_, 256), 256, 0, stream>>>(
      h2, sc2, sh2, h2b, BOT_, (size_t)B_ * BOT_);

  // --- decoder: h3 = relu(h2b @ dec_W^T + dec_b) -> bf16 directly ---
  gemm_wmma_bf16<EPI_RELU_BF16><<<dim3(B_ / 64, DEC_ / 32), 256, 0, stream>>>(
      h2b, decWb, dec_b, nullptr, h3b, nullptr, nullptr, B_, DEC_, BOT_);

  // --- output heads: fused activations, M_loss fused with M ---
  const dim3 headGrid(B_ / 64, INDIM / 32);  // 32 x 625
  gemm_wmma_bf16<EPI_SIGMOID><<<headGrid, 256, 0, stream>>>(
      h3b, o2Wb, o2_b, pai, nullptr, nullptr, nullptr, B_, INDIM, DEC_);
  gemm_wmma_bf16<EPI_EXP_MLOSS><<<headGrid, 256, 0, stream>>>(
      h3b, o1Wb, o1_b, Mout, nullptr, Mloss, s, B_, INDIM, DEC_);
  gemm_wmma_bf16<EPI_EXP><<<headGrid, 256, 0, stream>>>(
      h3b, o3Wb, o3_b, theta, nullptr, nullptr, nullptr, B_, INDIM, DEC_);
}
